// NeRFMultiResHashEncoder_5454608466294
// MI455X (gfx1250) — matrix operand order repaired and checked by reading, not compiled, not run on hardware
//
#include <hip/hip_runtime.h>
#include <hip/hip_bf16.h>
#include <stdint.h>

typedef __attribute__((ext_vector_type(16))) __bf16 v16bf;
typedef __attribute__((ext_vector_type(8)))  float  v8f;

#define LVLS   16
#define TSZ    65536
#define BRES   16
#define NFRQ   12
#define D_IN   168
#define K0PAD  192
#define HIDN   128
#define ROWS   128
#define XLDK   200   // padded LDS stride for X (16 rows -> 16 distinct banks)
#define HLDK   136   // padded LDS stride for H

// workspace layout in dwords; weights stored fragment-native:
//   frag = 16x32 bf16 B-tile = [32 lanes][8 dwords] contiguous (1 KB)
#define W0P_OFF   0          // 48 frags (6 ksteps x 8 ntiles)
#define WHP_OFF   12288      // 4 layers x 32 frags (4 x 8)
#define WOP_OFF   45056      // 4 frags  (4 ksteps x 1 ntile)
#define WTOTAL    46080

__device__ __forceinline__ uint16_t f2bf(float f) {
  union { float f; uint32_t u; } x; x.f = f;
  uint32_t r = x.u + 0x7FFFu + ((x.u >> 16) & 1u);   // round-to-nearest-even
  return (uint16_t)(r >> 16);
}
__device__ __forceinline__ uint32_t pack2(float lo, float hi) {
  return (uint32_t)f2bf(lo) | ((uint32_t)f2bf(hi) << 16);
}

union ABFrag { v16bf v; uint32_t u[8]; };

__device__ __forceinline__ v8f wmma_bf16(const ABFrag& a, const ABFrag& b, v8f c) {
  return __builtin_amdgcn_wmma_f32_16x16x32_bf16(false, a.v, false, b.v,
                                                 (short)0, c, false, false);
}

// A fragment (16x32 bf16) from LDS, rows m0..m0+15, K = k0..k0+31.
// ISA layout: lanes 0-15 -> M=lane, K pairs (2j,2j+1) in V0-3, 16+2j in V4-7;
// lanes 16-31 identical with K offset +8.  Two contiguous 16B chunks per lane.
__device__ __forceinline__ void load_a(ABFrag& a, const uint16_t* X, int ldk,
                                       int m0, int k0, int lane) {
  const uint16_t* p = X + (m0 + (lane & 15)) * ldk + k0 + ((lane >> 4) << 3);
#pragma unroll
  for (int j = 0; j < 8; ++j) {
    int off = ((j & 4) << 2) + ((j & 3) << 1);       // 0,2,4,6,16,18,20,22
    a.u[j] = *(const uint32_t*)(p + off);
  }
}

// B fragment from fragment-native packed weights: 2x global_load_b128 per lane.
__device__ __forceinline__ void load_b(ABFrag& b, const uint32_t* __restrict__ frags,
                                       int fragIdx, int lane) {
  const uint32_t* p = frags + ((size_t)fragIdx * 32 + lane) * 8;
  uint4 q0 = ((const uint4*)p)[0];
  uint4 q1 = ((const uint4*)p)[1];
  b.u[0] = q0.x; b.u[1] = q0.y; b.u[2] = q0.z; b.u[3] = q0.w;
  b.u[4] = q1.x; b.u[5] = q1.y; b.u[6] = q1.z; b.u[7] = q1.w;
}

// Map a fragment dword (lane, j) to the source (k, n) of the B tile at (ks, nt):
//   K = ks*32 + (lane>=16 ? 8 : 0) + (j>=4 ? 16 : 0) + 2*(j&3)   (pair K, K+1 packed)
//   N = nt*16 + (lane & 15)
__global__ void pack_weights_kernel(const float* __restrict__ W0,
                                    const float* __restrict__ Wh,
                                    const float* __restrict__ Wout,
                                    uint32_t* __restrict__ wp) {
  int i = blockIdx.x * blockDim.x + threadIdx.x;
  if (i >= WTOTAL) return;
  int lane = (i >> 3) & 31, j = i & 7;
  int kin = ((lane >> 4) << 3) + ((j & 4) << 2) + ((j & 3) << 1);  // K within tile (pair base)
  int n15 = lane & 15;
  if (i < WHP_OFF) {                                 // W0: 168x128 (K-pad to 192)
    int frag = i >> 8;                               // 0..47
    int ks = frag >> 3, nt = frag & 7;
    int k = ks * 32 + kin, n = nt * 16 + n15;
    float lo = (k     < D_IN) ? W0[(size_t)k * HIDN + n]       : 0.0f;
    float hi = (k + 1 < D_IN) ? W0[(size_t)(k + 1) * HIDN + n] : 0.0f;
    wp[i] = pack2(lo, hi);
  } else if (i < WOP_OFF) {                          // 4 hidden 128x128
    int r = i - WHP_OFF;
    int layer = r >> 13, fr = (r >> 8) & 31;
    int ks = fr >> 3, nt = fr & 7;
    int k = ks * 32 + kin, n = nt * 16 + n15;
    const float* Wl = Wh + (size_t)layer * HIDN * HIDN;
    wp[i] = pack2(Wl[(size_t)k * HIDN + n], Wl[(size_t)(k + 1) * HIDN + n]);
  } else {                                           // Wout: 128x4 (N-pad to 16)
    int r = i - WOP_OFF;
    int ks = r >> 8;
    int k = ks * 32 + kin;
    float lo = (n15 < 4) ? Wout[(size_t)k * 4 + n15]       : 0.0f;
    float hi = (n15 < 4) ? Wout[(size_t)(k + 1) * 4 + n15] : 0.0f;
    wp[i] = pack2(lo, hi);
  }
}

__global__ void __launch_bounds__(256)
nerf_fused_kernel(const float* __restrict__ coords,
                  const float* __restrict__ dirs,
                  const float* __restrict__ env,
                  const float* __restrict__ table,
                  const uint32_t* __restrict__ wp,
                  float* __restrict__ out) {
  __shared__ uint16_t Xs[ROWS * XLDK];          // 51200 B
  __shared__ uint16_t Hs[2][ROWS * HLDK];       // 69632 B  (CDNA5: 320KB/WG)

  const int tid  = threadIdx.x;
  const int base = blockIdx.x * ROWS;

  // ---------------- encode phase: 2 roles x 128 points ----------------
  {
    int r  = tid & 127;
    int pt = base + r;
    uint16_t* Xrow = &Xs[r * XLDK];
    if (tid < 128) {
      // role 0: all 16 hash-grid levels
      float cx = coords[3 * pt], cy = coords[3 * pt + 1], cz = coords[3 * pt + 2];
#pragma unroll
      for (int l = 0; l < LVLS; ++l) {
        int res = BRES << l;
        float sx = cx * (float)res, sy = cy * (float)res, sz = cz * (float)res;
        float fx = floorf(sx), fy = floorf(sy), fz = floorf(sz);
        float rx = sx - fx, ry = sy - fy, rz = sz - fz;
        int px = min(max((int)fx, 0), res - 1);
        int py = min(max((int)fy, 0), res - 1);
        int pz = min(max((int)fz, 0), res - 1);
        const float* tl = table + (size_t)l * TSZ * 2;
        float a0 = 0.f, a1 = 0.f;
#pragma unroll
        for (int c = 0; c < 8; ++c) {
          int ox = c & 1, oy = (c >> 1) & 1, oz = (c >> 2) & 1;
          uint32_t idx;
          if (l < 2) {  // (res+1)^3 <= T only for res = 16, 32
            idx = (uint32_t)((px + ox) + (res + 1) * ((py + oy) + (res + 1) * (pz + oz)));
          } else {
            idx = (((uint32_t)(px + ox)) ^
                   ((uint32_t)(py + oy) * 2654435761u) ^
                   ((uint32_t)(pz + oz) * 805459861u)) & 0xFFFFu;
          }
          float2 tv = *(const float2*)(tl + 2 * (size_t)idx);
          float w = (ox ? rx : 1.f - rx) * (oy ? ry : 1.f - ry) * (oz ? rz : 1.f - rz);
          a0 += tv.x * w; a1 += tv.y * w;
        }
        Xrow[2 * l] = f2bf(a0); Xrow[2 * l + 1] = f2bf(a1);
      }
    } else {
      // role 1: frequency encode + env copy + K pad
      float d0 = dirs[3 * pt], d1 = dirs[3 * pt + 1], d2 = dirs[3 * pt + 2];
#pragma unroll
      for (int f = 0; f < NFRQ; ++f) {
        float fr = (float)(1 << f) * 3.14159265358979323846f;
        float a;
        a = d0 * fr; Xrow[32 + f]      = f2bf(__sinf(a)); Xrow[32 + 12 + f] = f2bf(__cosf(a));
        a = d1 * fr; Xrow[32 + 24 + f] = f2bf(__sinf(a)); Xrow[32 + 36 + f] = f2bf(__cosf(a));
        a = d2 * fr; Xrow[32 + 48 + f] = f2bf(__sinf(a)); Xrow[32 + 60 + f] = f2bf(__cosf(a));
      }
      const float* e = env + (size_t)pt * 64;
#pragma unroll 8
      for (int q = 0; q < 64; ++q) Xrow[104 + q] = f2bf(e[q]);
#pragma unroll
      for (int q = D_IN; q < K0PAD; ++q) Xrow[q] = 0;
    }
  }
  __syncthreads();

  const int lane = tid & 31;
  const int wv   = tid >> 5;
  const int m0   = wv * 16;            // each wave owns one 16-row stripe

  // ---------------- layer 0: X[128x192] @ W0[192x128] -> H0 ----------------
  {
    ABFrag a[6];
#pragma unroll
    for (int ks = 0; ks < 6; ++ks) load_a(a[ks], Xs, XLDK, m0, ks * 32, lane);
    const uint32_t* Wf = wp + W0P_OFF;
    __builtin_prefetch(wp + WHP_OFF + lane * 16, 0, 1);   // warm first hidden layer
#pragma unroll
    for (int nt = 0; nt < 8; ++nt) {
      v8f c = {};
#pragma unroll
      for (int ks = 0; ks < 6; ++ks) {
        ABFrag b; load_b(b, Wf, ks * 8 + nt, lane);
        c = wmma_bf16(a[ks], b, c);
      }
#pragma unroll
      for (int v = 0; v < 8; ++v) {
        int m = m0 + v + ((lane >> 4) << 3);
        Hs[0][m * HLDK + nt * 16 + (lane & 15)] = f2bf(fmaxf(c[v], 0.f));
      }
    }
  }
  __syncthreads();

  // ---------------- 4 hidden layers: H[128x128] @ Wh[128x128] ----------------
  int cur = 0;
  for (int layer = 0; layer < 4; ++layer) {
    const uint32_t* Wf = wp + WHP_OFF + layer * 8192;
    __builtin_prefetch(wp + WHP_OFF + (layer + 1) * 8192 + lane * 16, 0, 1);
    ABFrag a[4];
#pragma unroll
    for (int ks = 0; ks < 4; ++ks) load_a(a[ks], Hs[cur], HLDK, m0, ks * 32, lane);
#pragma unroll
    for (int nt = 0; nt < 8; ++nt) {
      v8f c = {};
#pragma unroll
      for (int ks = 0; ks < 4; ++ks) {
        ABFrag b; load_b(b, Wf, ks * 8 + nt, lane);
        c = wmma_bf16(a[ks], b, c);
      }
#pragma unroll
      for (int v = 0; v < 8; ++v) {
        int m = m0 + v + ((lane >> 4) << 3);
        Hs[cur ^ 1][m * HLDK + nt * 16 + (lane & 15)] = f2bf(fmaxf(c[v], 0.f));
      }
    }
    __syncthreads();
    cur ^= 1;
  }

  // ---------------- output layer: H[128x128] @ Wout[128x16(pad)] ----------------
  {
    const uint32_t* Wf = wp + WOP_OFF;
    ABFrag a[4];
#pragma unroll
    for (int ks = 0; ks < 4; ++ks) load_a(a[ks], Hs[cur], HLDK, m0, ks * 32, lane);
    v8f c = {};
#pragma unroll
    for (int ks = 0; ks < 4; ++ks) {
      ABFrag b; load_b(b, Wf, ks, lane);
      c = wmma_bf16(a[ks], b, c);
    }
    int n = lane & 15;
    if (n < 4) {                        // divergence only after all WMMA issued
#pragma unroll
      for (int v = 0; v < 8; ++v) {
        int m = m0 + v + ((lane >> 4) << 3);
        out[(size_t)(base + m) * 4 + n] = c[v];
      }
    }
  }
}

extern "C" void kernel_launch(void* const* d_in, const int* in_sizes, int n_in,
                              void* d_out, int out_size, void* d_ws, size_t ws_size,
                              hipStream_t stream) {
  const float* coords = (const float*)d_in[0];
  const float* dirs   = (const float*)d_in[1];
  const float* env    = (const float*)d_in[2];
  const float* table  = (const float*)d_in[3];
  const float* W0     = (const float*)d_in[4];
  const float* Wh     = (const float*)d_in[5];
  const float* Wout   = (const float*)d_in[6];
  uint32_t* wp = (uint32_t*)d_ws;
  float* outp  = (float*)d_out;

  pack_weights_kernel<<<(WTOTAL + 255) / 256, 256, 0, stream>>>(W0, Wh, Wout, wp);

  int npts   = in_sizes[0] / 3;     // 524288
  int blocks = npts / ROWS;         // 4096 workgroups of 128 points
  nerf_fused_kernel<<<blocks, 256, 0, stream>>>(coords, dirs, env, table, wp, outp);
  (void)n_in; (void)out_size; (void)ws_size;
}
